// MultiHeadAttention_52089363366040
// MI455X (gfx1250) — matrix operand order repaired and checked by reading, not compiled, not run on hardware
//
#include <hip/hip_runtime.h>

// ---------------------------------------------------------------------------
// MultiHeadAttention for MI455X (gfx1250, wave32, WMMA + async-to-LDS).
//   B=4, S=2048, E=512, H=8, dh=64
// Pipeline: (1) QKV per-head 64x64 projections (f32 -> f16, WMMA f32 acc)
//           (2) flash-attention over 64-key blocks (WMMA QK^T and PV,
//               online softmax in f32, double-buffered LDS staging with
//               GLOBAL_LOAD_ASYNC_TO_LDS + s_wait_asynccnt)
//           (3) 512x512 output projection (WMMA, fp32 out)
// Workspace (f16): qp/kp/vp [B,H,S,64] (8MB each) + attn [B,S,E] (8MB) = 32MB
// ---------------------------------------------------------------------------

#define B_ 4
#define S_ 2048
#define E_ 512
#define H_ 8
#define DH 64

typedef __attribute__((ext_vector_type(16))) _Float16 v16h;
typedef __attribute__((ext_vector_type(8)))  _Float16 v8h;
typedef __attribute__((ext_vector_type(8)))  float    v8f;
typedef __attribute__((ext_vector_type(4)))  int      v4i;
typedef __attribute__((address_space(1)))    v4i      as1_v4i;  // global int4
typedef __attribute__((address_space(3)))    v4i      as3_v4i;  // LDS int4

// A-matrix fragment (16x32 f16). Row M = lane%16 (caller bakes row into ptr).
// lanes 0-15 hold K = {0..15}+k0 ; lanes 16-31 hold K = {16..31}+k0
__device__ __forceinline__ v16h ldA(const _Float16* row, int lhi) {
  v8h x = *(const v8h*)(row + lhi * 16);
  v8h y = *(const v8h*)(row + 8 + lhi * 16);
  v16h r;
#pragma unroll
  for (int i = 0; i < 8; ++i) { r[i] = x[i]; r[i + 8] = y[i]; }
  return r;
}

// B-matrix fragment (32x16 f16) from an "N-major" source: src[n][k] == B[k][n].
// Col N = lane%16, K = i + 16*(lane>=16). Caller passes &src[(n0+lane%16)*ld + k0 + lhi*16]
__device__ __forceinline__ v16h ldB(const _Float16* p) {
  return *(const v16h*)p;
}

__device__ __forceinline__ v8f wmma(v16h a, v16h b, v8f c) {
  return __builtin_amdgcn_wmma_f32_16x16x32_f16(false, a, false, b, (short)0, c,
                                                false, false);
}

// Async 16B global->LDS copy (per lane), tracked by ASYNCcnt.
__device__ __forceinline__ void async_cp_b128(const _Float16* g, _Float16* l) {
#if __has_builtin(__builtin_amdgcn_global_load_async_to_lds_b128)
  __builtin_amdgcn_global_load_async_to_lds_b128(
      (as1_v4i*)(g), (as3_v4i*)(l), 0, 0);
#else
  *(uint4*)l = *(const uint4*)g;   // fallback: regular vector copy via VGPRs
#endif
}

__device__ __forceinline__ void wait_async0() {
#if __has_builtin(__builtin_amdgcn_s_wait_asynccnt)
  __builtin_amdgcn_s_wait_asynccnt(0);
#else
  asm volatile("s_wait_asynccnt 0x0" ::: "memory");
#endif
}

// ---------------------------------------------------------------------------
// Kernel 1: per-head projection.  Rows m = (b*S+s)*H + h, x_row = X + m*64.
// out[o] = sum_d x[d] * W[o][d] + bias[o]   (torch y = x @ W.T + b)
// Output layout: [B,H,S,64] f16.
// ---------------------------------------------------------------------------
__global__ void __launch_bounds__(256) qkv_proj_kernel(
    const float* __restrict__ X, const float* __restrict__ W,
    const float* __restrict__ bias, _Float16* __restrict__ out) {
  __shared__ __align__(32) _Float16 Wl[DH * DH];     // W[o][k], N-major for ldB
  __shared__ __align__(32) _Float16 Xl[128 * DH];
  const int tid = threadIdx.x;
  const int lane = tid & 31, wave = tid >> 5;
  const int lrow = lane & 15, lhi = lane >> 4;
  const int m0 = blockIdx.x * 128;

  for (int i = tid; i < DH * DH; i += 256) Wl[i] = (_Float16)W[i];
  for (int i = tid; i < 128 * DH; i += 256) Xl[i] = (_Float16)X[(size_t)m0 * DH + i];
  __syncthreads();

  const _Float16* arow = &Xl[(wave * 16 + lrow) * DH];
  v16h a0 = ldA(arow, lhi);
  v16h a1 = ldA(arow + 32, lhi);

#pragma unroll
  for (int nt = 0; nt < 4; ++nt) {
    v8f acc;
    float bv = bias[nt * 16 + lrow];
#pragma unroll
    for (int r = 0; r < 8; ++r) acc[r] = bv;
    acc = wmma(a0, ldB(&Wl[(nt * 16 + lrow) * DH + lhi * 16]), acc);
    acc = wmma(a1, ldB(&Wl[(nt * 16 + lrow) * DH + 32 + lhi * 16]), acc);
#pragma unroll
    for (int r = 0; r < 8; ++r) {
      int m = m0 + wave * 16 + r + lhi * 8;
      int b = m >> 14;             // / (S*H)
      int s = (m >> 3) & (S_ - 1); // / H % S
      int h = m & (H_ - 1);
      out[(((size_t)(b * H_ + h) * S_) + s) * DH + nt * 16 + lrow] = (_Float16)acc[r];
    }
  }
}

// ---------------------------------------------------------------------------
// Kernel 2: flash attention for one (b,h); 128 query rows per block
// (8 waves x 16 rows); stream over 32 key-blocks of 64 keys.
// K blocks staged with async global->LDS copies, double-buffered so the next
// block's copy overlaps the current block's WMMAs.
// Writes combined heads: attn[b,s, h*64 + d]  (f16, [B,S,E]).
// ---------------------------------------------------------------------------
__global__ void __launch_bounds__(256) flash_attn_kernel(
    const _Float16* __restrict__ qp, const _Float16* __restrict__ kp,
    const _Float16* __restrict__ vp, _Float16* __restrict__ attn) {
  __shared__ __align__(32) _Float16 Kl[2][64 * DH];   // K[key][d]  (N-major, async)
  __shared__ __align__(32) _Float16 Vt[2][DH * 64];   // V^T[d][key] (N-major)
  __shared__ __align__(32) _Float16 Pl[8][16 * 64];   // per-wave P (C->A relayout)

  const int tid = threadIdx.x;
  const int lane = tid & 31, wave = tid >> 5;
  const int lrow = lane & 15, lhi = lane >> 4;
  const int bh = blockIdx.y;              // b*H + h
  const int b = bh >> 3, h = bh & 7;
  const size_t hoff = (size_t)bh * S_ * DH;
  const int q0 = blockIdx.x * 128 + wave * 16;

  const _Float16* qrow = qp + hoff + (size_t)(q0 + lrow) * DH;
  v16h aq0 = ldA(qrow, lhi);
  v16h aq1 = ldA(qrow + 32, lhi);

  // stage key-block `blk` into buffer `p`: K via async b128, V^T scalar
  auto stage = [&](int p, int blk) {
    const _Float16* kblk = kp + hoff + (size_t)blk * 64 * DH;
    const _Float16* vblk = vp + hoff + (size_t)blk * 64 * DH;
#pragma unroll
    for (int i = tid; i < 512; i += 256)
      async_cp_b128(kblk + i * 8, &Kl[p][i * 8]);
    for (int i = tid; i < 64 * DH; i += 256) {
      int key = i >> 6, d = i & 63;
      Vt[p][d * 64 + key] = vblk[i];
    }
  };

  float rm[8], rl[8];
  v8f oacc[4];
#pragma unroll
  for (int r = 0; r < 8; ++r) { rm[r] = -1e30f; rl[r] = 0.0f; }
#pragma unroll
  for (int nt = 0; nt < 4; ++nt)
#pragma unroll
    for (int r = 0; r < 8; ++r) oacc[nt][r] = 0.0f;

  stage(0, 0);
  wait_async0();
  __syncthreads();

  for (int kb = 0; kb < S_ / 64; ++kb) {
    const int p = kb & 1;
    if (kb + 1 < S_ / 64) stage(p ^ 1, kb + 1);   // overlap with compute below

    // scores = q @ K^T * 1/sqrt(dh)   (16 x 64 per wave)
    v8f sc[4];
#pragma unroll
    for (int nt = 0; nt < 4; ++nt) {
      v8f z;
#pragma unroll
      for (int r = 0; r < 8; ++r) z[r] = 0.0f;
      z = wmma(aq0, ldB(&Kl[p][(nt * 16 + lrow) * DH + lhi * 16]), z);
      z = wmma(aq1, ldB(&Kl[p][(nt * 16 + lrow) * DH + 32 + lhi * 16]), z);
#pragma unroll
      for (int r = 0; r < 8; ++r) sc[nt][r] = z[r] * 0.125f;
    }

    // online softmax: row stats shared across the 16-lane N group
    float nm[8];
#pragma unroll
    for (int r = 0; r < 8; ++r) {
      float v = rm[r];
#pragma unroll
      for (int nt = 0; nt < 4; ++nt) v = fmaxf(v, sc[nt][r]);
      nm[r] = v;
    }
#pragma unroll
    for (int mask = 1; mask <= 8; mask <<= 1)
#pragma unroll
      for (int r = 0; r < 8; ++r)
        nm[r] = fmaxf(nm[r], __shfl_xor(nm[r], mask, 32));

    float sf[8], rs[8];
#pragma unroll
    for (int r = 0; r < 8; ++r) {
      sf[r] = __expf(rm[r] - nm[r]);
      rm[r] = nm[r];
      rs[r] = 0.0f;
    }
#pragma unroll
    for (int nt = 0; nt < 4; ++nt)
#pragma unroll
      for (int r = 0; r < 8; ++r) {
        float pv = __expf(sc[nt][r] - nm[r]);
        sc[nt][r] = pv;
        rs[r] += pv;
      }
#pragma unroll
    for (int mask = 1; mask <= 8; mask <<= 1)
#pragma unroll
      for (int r = 0; r < 8; ++r)
        rs[r] += __shfl_xor(rs[r], mask, 32);
#pragma unroll
    for (int r = 0; r < 8; ++r) rl[r] = rl[r] * sf[r] + rs[r];
#pragma unroll
    for (int nt = 0; nt < 4; ++nt)
#pragma unroll
      for (int r = 0; r < 8; ++r) oacc[nt][r] *= sf[r];

    // P: C-layout -> per-wave LDS -> A-layout (wave-local, DS ops in order)
    _Float16* pw = &Pl[wave][0];
#pragma unroll
    for (int nt = 0; nt < 4; ++nt)
#pragma unroll
      for (int r = 0; r < 8; ++r)
        pw[(r + lhi * 8) * 64 + nt * 16 + lrow] = (_Float16)sc[nt][r];

    // O += P @ V
#pragma unroll
    for (int kt = 0; kt < 2; ++kt) {
      v16h ap = ldA(pw + lrow * 64 + kt * 32, lhi);
#pragma unroll
      for (int nt = 0; nt < 4; ++nt)
        oacc[nt] = wmma(ap, ldB(&Vt[p][(nt * 16 + lrow) * 64 + kt * 32 + lhi * 16]),
                        oacc[nt]);
    }

    wait_async0();      // next block's async K copy has landed in LDS
    __syncthreads();    // all waves done with buffer p; safe to reuse next iter
  }

  // normalize + write combined heads [B,S,E] f16
#pragma unroll
  for (int nt = 0; nt < 4; ++nt)
#pragma unroll
    for (int r = 0; r < 8; ++r) {
      int s = q0 + r + lhi * 8;
      float o = oacc[nt][r] / rl[r];
      attn[((size_t)(b * S_ + s)) * E_ + h * DH + nt * 16 + lrow] = (_Float16)o;
    }
}

// ---------------------------------------------------------------------------
// Kernel 3: output projection.  out[m][n] = sum_k attn[m][k]*Wo[n][k] + b[n]
// M=8192, N=512, K=512. Block: 128 rows x 64 cols; Wo tile staged f16 in LDS.
// ---------------------------------------------------------------------------
__global__ void __launch_bounds__(256) out_proj_kernel(
    const _Float16* __restrict__ attn, const float* __restrict__ Wo,
    const float* __restrict__ bias, float* __restrict__ out) {
  __shared__ __align__(32) _Float16 Wl[64 * 64];   // Wo[n0+nn][k0+kk] (N-major)
  const int tid = threadIdx.x;
  const int lane = tid & 31, wave = tid >> 5;
  const int lrow = lane & 15, lhi = lane >> 4;
  const int m0 = blockIdx.x * 128;
  const int n0 = blockIdx.y * 64;

  v8f acc[4];
#pragma unroll
  for (int nt = 0; nt < 4; ++nt) {
    float bv = bias[n0 + nt * 16 + lrow];
#pragma unroll
    for (int r = 0; r < 8; ++r) acc[nt][r] = bv;
  }

  const _Float16* arow = attn + (size_t)(m0 + wave * 16 + lrow) * E_;

  for (int kc = 0; kc < E_ / 64; ++kc) {
    const int k0 = kc * 64;
    for (int i = tid; i < 64 * 64; i += 256) {
      int nn = i >> 6, kk = i & 63;
      Wl[i] = (_Float16)Wo[(size_t)(n0 + nn) * E_ + k0 + kk];
    }
    __syncthreads();
    v16h a0 = ldA(arow + k0, lhi);
    v16h a1 = ldA(arow + k0 + 32, lhi);
#pragma unroll
    for (int nt = 0; nt < 4; ++nt) {
      acc[nt] = wmma(a0, ldB(&Wl[(nt * 16 + lrow) * 64 + lhi * 16]), acc[nt]);
      acc[nt] = wmma(a1, ldB(&Wl[(nt * 16 + lrow) * 64 + 32 + lhi * 16]), acc[nt]);
    }
    __syncthreads();
  }

#pragma unroll
  for (int nt = 0; nt < 4; ++nt)
#pragma unroll
    for (int r = 0; r < 8; ++r) {
      int m = m0 + wave * 16 + r + lhi * 8;
      out[(size_t)m * E_ + n0 + nt * 16 + lrow] = acc[nt][r];
    }
}

// ---------------------------------------------------------------------------
extern "C" void kernel_launch(void* const* d_in, const int* in_sizes, int n_in,
                              void* d_out, int out_size, void* d_ws, size_t ws_size,
                              hipStream_t stream) {
  const float* Q    = (const float*)d_in[0];
  const float* K    = (const float*)d_in[1];
  const float* V    = (const float*)d_in[2];
  const float* Wq_w = (const float*)d_in[3];
  const float* Wq_b = (const float*)d_in[4];
  const float* Wk_w = (const float*)d_in[5];
  const float* Wk_b = (const float*)d_in[6];
  const float* Wv_w = (const float*)d_in[7];
  const float* Wv_b = (const float*)d_in[8];
  const float* Wo_w = (const float*)d_in[9];
  const float* Wo_b = (const float*)d_in[10];
  float* out = (float*)d_out;

  const size_t HEAD_ELEMS = (size_t)B_ * H_ * S_ * DH;   // 4,194,304
  _Float16* ws = (_Float16*)d_ws;
  _Float16* qp = ws;
  _Float16* kp = ws + HEAD_ELEMS;
  _Float16* vp = ws + 2 * HEAD_ELEMS;
  _Float16* at = ws + 3 * HEAD_ELEMS;                    // [B,S,E] f16

  const int PROJ_BLOCKS = (B_ * S_ * H_) / 128;          // 512
  qkv_proj_kernel<<<PROJ_BLOCKS, 256, 0, stream>>>(Q, Wq_w, Wq_b, qp);
  qkv_proj_kernel<<<PROJ_BLOCKS, 256, 0, stream>>>(K, Wk_w, Wk_b, kp);
  qkv_proj_kernel<<<PROJ_BLOCKS, 256, 0, stream>>>(V, Wv_w, Wv_b, vp);

  flash_attn_kernel<<<dim3(S_ / 128, B_ * H_), 256, 0, stream>>>(qp, kp, vp, at);

  out_proj_kernel<<<dim3((B_ * S_) / 128, E_ / 64), 256, 0, stream>>>(at, Wo_w,
                                                                      Wo_b, out);
}
